// HybridModel_28733331210932
// MI455X (gfx1250) — matrix operand order repaired and checked by reading, not compiled, not run on hardware
//
#include <hip/hip_runtime.h>
#include <hip/hip_bf16.h>

typedef __attribute__((ext_vector_type(16))) _Float16 v16h;
typedef __attribute__((ext_vector_type(8)))  _Float16 v8h;
typedef __attribute__((ext_vector_type(8)))  float    v8f;

#define NNODES 16384
#define NEDGES 524288
#define ETOT   (NEDGES + NNODES)   // self loops appended
#define NEG_SLOPE 0.2f

// ---------------- order-preserving float <-> int key for atomicMax ----------
__device__ __forceinline__ int fkey(float f) {
    int b = __float_as_int(f);
    return b >= 0 ? b : (b ^ 0x7fffffff);
}
__device__ __forceinline__ float funkey(int k) {
    int b = k >= 0 ? k : (k ^ 0x7fffffff);
    return __int_as_float(b);
}

// ---------------- fused CNN: conv1+relu+pool, conv2+relu+pool, per image ----
__global__ __launch_bounds__(256) void cnn_kernel(
    const float* __restrict__ x,      // [N,1,28,28]
    const float* __restrict__ c1w, const float* __restrict__ c1b,
    const float* __restrict__ c2w, const float* __restrict__ c2b,
    _Float16* __restrict__ h2)        // [N,1568] = [N,32,7,7] flattened, f16
{
    __shared__ float sIn[28 * 28];
    __shared__ float sP1[16 * 14 * 14];
    __shared__ float sW1[16 * 9];
    __shared__ float sB1[16];
    __shared__ float sW2[32 * 16 * 9];
    __shared__ float sB2[32];
    const int t = threadIdx.x;
    const int n = blockIdx.x;

    for (int i = t; i < 16 * 9; i += 256) sW1[i] = c1w[i];
    if (t < 16) sB1[t] = c1b[t];
    for (int i = t; i < 32 * 16 * 9; i += 256) sW2[i] = c2w[i];
    if (t < 32) sB2[t] = c2b[t];
    const float* xin = x + (long)n * 784;
    for (int i = t; i < 784; i += 256) sIn[i] = xin[i];
    __syncthreads();

    // conv1 (SAME) + relu + 2x2 maxpool -> sP1 [16,14,14]
    for (int i = t; i < 16 * 14 * 14; i += 256) {
        int oc = i / 196, r = i % 196, py = r / 14, px = r % 14;
        const float* w = sW1 + oc * 9;
        float mx = 0.f;  // relu outputs are >=0, so 0 is a valid identity
        for (int dy = 0; dy < 2; ++dy)
        for (int dx = 0; dx < 2; ++dx) {
            int cy = py * 2 + dy, cx = px * 2 + dx;
            float acc = sB1[oc];
            for (int ky = 0; ky < 3; ++ky) {
                int iy = cy + ky - 1;
                if ((unsigned)iy < 28u)
                    for (int kx = 0; kx < 3; ++kx) {
                        int ix = cx + kx - 1;
                        if ((unsigned)ix < 28u) acc += sIn[iy * 28 + ix] * w[ky * 3 + kx];
                    }
            }
            float rv = acc > 0.f ? acc : 0.f;
            mx = mx > rv ? mx : rv;
        }
        sP1[i] = mx;
    }
    __syncthreads();

    // conv2 (SAME, 16ch) + relu + 2x2 maxpool -> h2 [32,7,7] f16
    _Float16* out = h2 + (long)n * 1568;
    for (int i = t; i < 32 * 7 * 7; i += 256) {
        int oc = i / 49, r = i % 49, py = r / 7, px = r % 7;
        const float* w = sW2 + oc * 144;
        float mx = 0.f;
        for (int dy = 0; dy < 2; ++dy)
        for (int dx = 0; dx < 2; ++dx) {
            int cy = py * 2 + dy, cx = px * 2 + dx;
            float acc = sB2[oc];
            for (int ic = 0; ic < 16; ++ic) {
                const float* pin = sP1 + ic * 196;
                const float* wk  = w + ic * 9;
                for (int ky = 0; ky < 3; ++ky) {
                    int iy = cy + ky - 1;
                    if ((unsigned)iy < 14u)
                        for (int kx = 0; kx < 3; ++kx) {
                            int ix = cx + kx - 1;
                            if ((unsigned)ix < 14u) acc += pin[iy * 14 + ix] * wk[ky * 3 + kx];
                        }
                }
            }
            float rv = acc > 0.f ? acc : 0.f;
            mx = mx > rv ? mx : rv;
        }
        out[i] = (_Float16)mx;
    }
}

// ---------------- weight convert + transpose: W[K,M] f32 -> Wt[M,K] f16 -----
__global__ void convertWT_kernel(const float* __restrict__ W, _Float16* __restrict__ Wt,
                                 int K, int M) {
    int idx = blockIdx.x * blockDim.x + threadIdx.x;
    if (idx < K * M) {
        int k = idx / M, m = idx % M;
        Wt[(long)m * K + k] = (_Float16)W[idx];
    }
}

// ---------------- WMMA GEMM: C[N,MOUT] = A[N,K](f16) x Wt[MOUT,K](f16) + bias
// One wave computes a 16x16 tile. 8 waves / block. Grids sized exactly.
template <int K, int MOUT>
__global__ __launch_bounds__(256) void gemm_wmma_kernel(
    const _Float16* __restrict__ A,   // [N, K] row-major f16
    const _Float16* __restrict__ Bt,  // [MOUT, K] row-major f16 (transposed W)
    const float* __restrict__ bias,   // [MOUT]
    float* __restrict__ C)            // [N, MOUT] f32
{
    const int lane = threadIdx.x & 31;
    const int wave = threadIdx.x >> 5;
    const int tile = blockIdx.x * 8 + wave;
    const int colTiles = MOUT / 16;
    const int rowTile = tile / colTiles;
    const int colTile = tile % colTiles;

    const int m   = lane & 15;
    const int grp = (lane >> 4) & 1;
    const long arow = (long)(rowTile * 16 + m) * K;
    const long brow = (long)(colTile * 16 + m) * K;

    v8f acc = {};
    for (int k0 = 0; k0 < K; k0 += 32) {
        // A fragment: halves 0..7 -> k0 + grp*8 + i ; halves 8..15 -> +16
        const _Float16* ap = A + arow + k0 + grp * 8;
        v8h alo = *(const v8h*)(ap);
        v8h ahi = *(const v8h*)(ap + 16);
        // B fragment: halves 0..7 -> k0 + grp*8 + i ; halves 8..15 -> +16
        const _Float16* bp = Bt + brow + k0 + grp * 8;
        v8h blo = *(const v8h*)(bp);
        v8h bhi = *(const v8h*)(bp + 16);
        if (k0 + 32 < K) {
            __builtin_prefetch(ap + 32, 0, 3);
            __builtin_prefetch(bp + 32, 0, 3);
        }
        v16h a, b;
#pragma unroll
        for (int i = 0; i < 8; ++i) {
            a[i] = alo[i]; a[i + 8] = ahi[i];
            b[i] = blo[i]; b[i + 8] = bhi[i];
        }
        acc = __builtin_amdgcn_wmma_f32_16x16x32_f16(
            false, a, false, b, (short)0, acc, false, false);
    }
    // C/D layout: VGPR v, lane L -> row = v + 8*(L>=16), col = L&15
#pragma unroll
    for (int v = 0; v < 8; ++v) {
        int row = rowTile * 16 + v + 8 * grp;
        int col = colTile * 16 + m;
        C[(long)row * MOUT + col] = acc[v] + bias[col];
    }
}

// ---------------- fills ------------------------------------------------------
__global__ void fill_f32_kernel(float* p, float v, int n) {
    int i = blockIdx.x * blockDim.x + threadIdx.x;
    if (i < n) p[i] = v;
}
__global__ void fill_i32_kernel(int* p, int v, int n) {
    int i = blockIdx.x * blockDim.x + threadIdx.x;
    if (i < n) p[i] = v;
}

// ---------------- GATv2 edge kernels (C=64 channels per head) ----------------
template <int H>
__global__ void att_alpha_kernel(const float* __restrict__ xl, const float* __restrict__ xr,
                                 const float* __restrict__ att, const int* __restrict__ ei,
                                 float* __restrict__ alpha, int* __restrict__ amaxKey) {
    int idx = blockIdx.x * blockDim.x + threadIdx.x;
    if (idx >= ETOT * H) return;
    int e = idx / H, h = idx % H;
    int s, d;
    if (e < NEDGES) { s = ei[e]; d = ei[NEDGES + e]; } else { s = d = e - NEDGES; }
    const float* pl = xl + (long)s * (H * 64) + h * 64;
    const float* pr = xr + (long)d * (H * 64) + h * 64;
    const float* pa = att + h * 64;
    float acc = 0.f;
#pragma unroll 8
    for (int c = 0; c < 64; ++c) {
        float v = pl[c] + pr[c];
        v = v > 0.f ? v : NEG_SLOPE * v;
        acc += v * pa[c];
    }
    alpha[idx] = acc;
    atomicMax(&amaxKey[d * H + h], fkey(acc));
}

template <int H>
__global__ void att_exp_kernel(const int* __restrict__ ei, float* __restrict__ alpha,
                               const int* __restrict__ amaxKey, float* __restrict__ den) {
    int idx = blockIdx.x * blockDim.x + threadIdx.x;
    if (idx >= ETOT * H) return;
    int e = idx / H, h = idx % H;
    int d;
    if (e < NEDGES) d = ei[NEDGES + e]; else d = e - NEDGES;
    float mx = funkey(amaxKey[d * H + h]);
    float ex = __expf(alpha[idx] - mx);
    alpha[idx] = ex;
    atomicAdd(&den[d * H + h], ex);
}

template <int H>
__global__ void att_scatter_kernel(const int* __restrict__ ei, const float* __restrict__ alpha,
                                   const float* __restrict__ den, const float* __restrict__ xl,
                                   float* __restrict__ out) {
    int idx = blockIdx.x * blockDim.x + threadIdx.x;
    if (idx >= ETOT * H) return;
    int e = idx / H, h = idx % H;
    int s, d;
    if (e < NEDGES) { s = ei[e]; d = ei[NEDGES + e]; } else { s = d = e - NEDGES; }
    float w = alpha[idx] / den[d * H + h];
    const float* pl = xl + (long)s * (H * 64) + h * 64;
    float* po = out + (long)d * (H * 64) + h * 64;
#pragma unroll 8
    for (int c = 0; c < 64; ++c) atomicAdd(&po[c], pl[c] * w);
}

// ---------------- bias + relu + f16 convert ---------------------------------
__global__ void bias_relu_f16_kernel(const float* __restrict__ acc, const float* __restrict__ bias,
                                     _Float16* __restrict__ out, int n, int m) {
    int i = blockIdx.x * blockDim.x + threadIdx.x;
    if (i < n * m) {
        float v = acc[i] + bias[i % m];
        v = v > 0.f ? v : 0.f;
        out[i] = (_Float16)v;
    }
}

// ---------------- classifier: (h+bias2) @ Wc + bc ---------------------------
__global__ void classifier_kernel(const float* __restrict__ h, const float* __restrict__ bias2,
                                  const float* __restrict__ Wc, const float* __restrict__ bc,
                                  float* __restrict__ out) {
    int i = blockIdx.x * blockDim.x + threadIdx.x;
    if (i >= NNODES * 10) return;
    int n = i / 10, j = i % 10;
    const float* ph = h + (long)n * 64;
    float acc = bc[j];
#pragma unroll 8
    for (int c = 0; c < 64; ++c) acc += (ph[c] + bias2[c]) * Wc[c * 10 + j];
    out[i] = acc;
}

extern "C" void kernel_launch(void* const* d_in, const int* in_sizes, int n_in,
                              void* d_out, int out_size, void* d_ws, size_t ws_size,
                              hipStream_t stream) {
    (void)in_sizes; (void)n_in; (void)out_size; (void)ws_size;
    const float* x       = (const float*)d_in[0];
    const int*   ei      = (const int*)d_in[1];
    const float* c1w     = (const float*)d_in[2];
    const float* c1b     = (const float*)d_in[3];
    const float* c2w     = (const float*)d_in[4];
    const float* c2b     = (const float*)d_in[5];
    const float* Wl1     = (const float*)d_in[6];
    const float* bl1     = (const float*)d_in[7];
    const float* Wr1     = (const float*)d_in[8];
    const float* br1     = (const float*)d_in[9];
    const float* att1    = (const float*)d_in[10];
    const float* bias1   = (const float*)d_in[11];
    const float* Wl2     = (const float*)d_in[12];
    const float* bl2     = (const float*)d_in[13];
    const float* Wr2     = (const float*)d_in[14];
    const float* br2     = (const float*)d_in[15];
    const float* att2    = (const float*)d_in[16];
    const float* bias2   = (const float*)d_in[17];
    const float* Wc      = (const float*)d_in[18];
    const float* bc      = (const float*)d_in[19];
    float* outp = (float*)d_out;

    size_t ofs = 0;
    auto alloc = [&](size_t bytes) -> void* {
        void* p = (char*)d_ws + ofs;
        ofs += (bytes + 255) & ~(size_t)255;
        return p;
    };
    _Float16* h2f16  = (_Float16*)alloc((size_t)NNODES * 1568 * 2);
    _Float16* WtL1   = (_Float16*)alloc((size_t)128 * 1568 * 2);
    _Float16* WtR1   = (_Float16*)alloc((size_t)128 * 1568 * 2);
    _Float16* WtL2   = (_Float16*)alloc((size_t)64 * 128 * 2);
    _Float16* WtR2   = (_Float16*)alloc((size_t)64 * 128 * 2);
    float* xl1  = (float*)alloc((size_t)NNODES * 128 * 4);
    float* xr1  = (float*)alloc((size_t)NNODES * 128 * 4);
    float* al1  = (float*)alloc((size_t)ETOT * 2 * 4);
    int*   mk1  = (int*)  alloc((size_t)NNODES * 2 * 4);
    float* dn1  = (float*)alloc((size_t)NNODES * 2 * 4);
    float* out1 = (float*)alloc((size_t)NNODES * 128 * 4);
    _Float16* h1f16 = (_Float16*)alloc((size_t)NNODES * 128 * 2);
    float* xl2  = (float*)alloc((size_t)NNODES * 64 * 4);
    float* xr2  = (float*)alloc((size_t)NNODES * 64 * 4);
    float* al2  = (float*)alloc((size_t)ETOT * 4);
    int*   mk2  = (int*)  alloc((size_t)NNODES * 4);
    float* dn2  = (float*)alloc((size_t)NNODES * 4);
    float* out2 = (float*)alloc((size_t)NNODES * 64 * 4);

    // 1) CNN feature extractor -> h2f16 [N,1568]
    cnn_kernel<<<NNODES, 256, 0, stream>>>(x, c1w, c1b, c2w, c2b, h2f16);

    // 2) weight convert+transpose to f16
    convertWT_kernel<<<(1568 * 128 + 255) / 256, 256, 0, stream>>>(Wl1, WtL1, 1568, 128);
    convertWT_kernel<<<(1568 * 128 + 255) / 256, 256, 0, stream>>>(Wr1, WtR1, 1568, 128);
    convertWT_kernel<<<(128 * 64 + 255) / 256, 256, 0, stream>>>(Wl2, WtL2, 128, 64);
    convertWT_kernel<<<(128 * 64 + 255) / 256, 256, 0, stream>>>(Wr2, WtR2, 128, 64);

    // 3) GAT layer 1 linears (WMMA): [N,1568] x [1568,128]
    gemm_wmma_kernel<1568, 128><<<1024, 256, 0, stream>>>(h2f16, WtL1, bl1, xl1);
    gemm_wmma_kernel<1568, 128><<<1024, 256, 0, stream>>>(h2f16, WtR1, br1, xr1);

    // 4) layer-1 edge softmax + scatter (H=2, C=64)
    fill_f32_kernel<<<(NNODES * 128 + 255) / 256, 256, 0, stream>>>(out1, 0.f, NNODES * 128);
    fill_f32_kernel<<<(NNODES * 2 + 255) / 256, 256, 0, stream>>>(dn1, 0.f, NNODES * 2);
    fill_i32_kernel<<<(NNODES * 2 + 255) / 256, 256, 0, stream>>>(mk1, INT_MIN, NNODES * 2);
    {
        int th = ETOT * 2, blocks = (th + 255) / 256;
        att_alpha_kernel<2><<<blocks, 256, 0, stream>>>(xl1, xr1, att1, ei, al1, mk1);
        att_exp_kernel<2><<<blocks, 256, 0, stream>>>(ei, al1, mk1, dn1);
        att_scatter_kernel<2><<<blocks, 256, 0, stream>>>(ei, al1, dn1, xl1, out1);
    }
    // 5) +bias1, relu, convert to f16 for layer-2 GEMMs
    bias_relu_f16_kernel<<<(NNODES * 128 + 255) / 256, 256, 0, stream>>>(out1, bias1, h1f16,
                                                                        NNODES, 128);

    // 6) GAT layer 2 linears (WMMA): [N,128] x [128,64]
    gemm_wmma_kernel<128, 64><<<512, 256, 0, stream>>>(h1f16, WtL2, bl2, xl2);
    gemm_wmma_kernel<128, 64><<<512, 256, 0, stream>>>(h1f16, WtR2, br2, xr2);

    // 7) layer-2 edge softmax + scatter (H=1, C=64)
    fill_f32_kernel<<<(NNODES * 64 + 255) / 256, 256, 0, stream>>>(out2, 0.f, NNODES * 64);
    fill_f32_kernel<<<(NNODES + 255) / 256, 256, 0, stream>>>(dn2, 0.f, NNODES);
    fill_i32_kernel<<<(NNODES + 255) / 256, 256, 0, stream>>>(mk2, INT_MIN, NNODES);
    {
        int th = ETOT, blocks = (th + 255) / 256;
        att_alpha_kernel<1><<<blocks, 256, 0, stream>>>(xl2, xr2, att2, ei, al2, mk2);
        att_exp_kernel<1><<<blocks, 256, 0, stream>>>(ei, al2, mk2, dn2);
        att_scatter_kernel<1><<<blocks, 256, 0, stream>>>(ei, al2, dn2, xl2, out2);
    }

    // 8) classifier: (out2 + bias2) @ Wc + bc -> d_out [N,10]
    classifier_kernel<<<(NNODES * 10 + 255) / 256, 256, 0, stream>>>(out2, bias2, Wc, bc, outp);
}
